// prob_calc_added_layers_TSP_64226940944671
// MI455X (gfx1250) — matrix operand order repaired and checked by reading, not compiled, not run on hardware
//
#include <hip/hip_runtime.h>

// MI455X / gfx1250, wave32. All GEMMs via v_wmma_f32_16x16x32_f16 (f32 accum).
// d_ws layout (f16): Q[B][112][128] | K[B][112][128] | VT[B][128][128]

typedef __attribute__((ext_vector_type(16))) _Float16 v16h;
typedef __attribute__((ext_vector_type(8)))  float    v8f;

#define BATCH_ 1024
#define EMB_   128
#define NROW_  100      // N == G == 100
#define NPAD_  112      // 7 tiles of 16
#define MT_    7

__device__ static inline int lane_id() { return (int)(threadIdx.x & 31u); }

__device__ static inline v8f wmma16(v16h a, v16h b, v8f c) {
  return __builtin_amdgcn_wmma_f32_16x16x32_f16(false, a, false, b, (short)0, c, false, false);
}

// ---- A fragment 16x32 (f16) from f32 row-major src, rows >= nrows give 0 ----
// element(m = lane&15, k): elems 0..15 -> k = kb+j ; kb = (lane>>4)*16
__device__ static inline v16h frag_a_f32(const float* __restrict__ src, int ld,
                                         int rowbase, int nrows, int koff) {
  int l = lane_id(); int m = l & 15; int kb = (l >> 4) * 16;
  int row = rowbase + m;
  v16h a;
  if (row < nrows) {
    const float* p = src + row * ld + koff + kb;
#pragma unroll
    for (int j = 0; j < 8; ++j) a[j] = (_Float16)p[j];
#pragma unroll
    for (int j = 0; j < 8; ++j) a[8 + j] = (_Float16)p[8 + j];
  } else {
#pragma unroll
    for (int j = 0; j < 16; ++j) a[j] = (_Float16)0.0f;
  }
  return a;
}

// ---- A fragment from f16 LDS row-major (rows pre-padded/zeroed) ----
__device__ static inline v16h frag_a_lds(const _Float16* src, int ld, int rowbase, int koff) {
  int l = lane_id(); int m = l & 15; int kb = (l >> 4) * 16;
  const _Float16* p = src + (rowbase + m) * ld + koff + kb;
  v16h a;
#pragma unroll
  for (int j = 0; j < 8; ++j) a[j] = p[j];
#pragma unroll
  for (int j = 0; j < 8; ++j) a[8 + j] = p[8 + j];
  return a;
}

// ---- B fragment 32x16: element(k,n) = src[n][k]  (X @ W^T pattern, f32 src) ----
// lane: n = nbase + (lane&15), k = koff + (lane>>4)*16 + j (16 contiguous)
__device__ static inline v16h frag_b_wT_f32(const float* __restrict__ src, int ld,
                                            int nbase, int koff) {
  int l = lane_id();
  const float* p = src + (nbase + (l & 15)) * ld + koff + ((l >> 4) * 16);
  v16h b;
#pragma unroll
  for (int j = 0; j < 16; ++j) b[j] = (_Float16)p[j];
  return b;
}

// same, but rows (n) >= nrows give zeros (for nodes with 100 rows)
__device__ static inline v16h frag_b_wT_f32_guard(const float* __restrict__ src, int ld,
                                                  int nbase, int nrows, int koff) {
  int l = lane_id(); int n = nbase + (l & 15);
  v16h b;
  if (n < nrows) {
    const float* p = src + n * ld + koff + ((l >> 4) * 16);
#pragma unroll
    for (int j = 0; j < 16; ++j) b[j] = (_Float16)p[j];
  } else {
#pragma unroll
    for (int j = 0; j < 16; ++j) b[j] = (_Float16)0.0f;
  }
  return b;
}

// ---- B fragment: element(k,n) = src[k][n]  (row-major B, e.g. new[b]) ----
__device__ static inline v16h frag_b_rm_f32(const float* __restrict__ src, int ld,
                                            int nbase, int koff) {
  int l = lane_id(); int n = nbase + (l & 15); int k0 = koff + (l >> 4) * 16;
  v16h b;
#pragma unroll
  for (int j = 0; j < 16; ++j) b[j] = (_Float16)src[(k0 + j) * ld + n];
  return b;
}

// ---- B fragment f16: element(k,n) = src[n][k]  (VT: emb-major V) ----
__device__ static inline v16h frag_b_wT_f16(const _Float16* __restrict__ src, int ld,
                                            int nbase, int koff) {
  int l = lane_id();
  const _Float16* p = src + (nbase + (l & 15)) * ld + koff + ((l >> 4) * 16);
  v16h b;
#pragma unroll
  for (int j = 0; j < 16; ++j) b[j] = p[j];
  return b;
}

// ---- B fragment for scores: element(k,n) = (k<16) ? K[n][hbase+k] : 0 ----
__device__ static inline v16h frag_b_khead(const _Float16* __restrict__ K,
                                           int nbase, int hbase) {
  int l = lane_id();
  v16h b;
  if (l < 16) {
    const _Float16* p = K + (nbase + l) * EMB_ + hbase;
#pragma unroll
    for (int j = 0; j < 16; ++j) b[j] = p[j];
  } else {
#pragma unroll
    for (int j = 0; j < 16; ++j) b[j] = (_Float16)0.0f;
  }
  return b;
}

// ---- A fragment for Q head slice: k in [0,16) real, [16,32) zero ----
__device__ static inline v16h frag_a_qhead(const _Float16* __restrict__ Q,
                                           int rowbase, int hbase) {
  int l = lane_id(); int m = l & 15; int kb = (l >> 4) * 16;
  const _Float16* p = Q + (rowbase + m) * EMB_ + hbase;
  v16h a;
#pragma unroll
  for (int j = 0; j < 16; ++j) a[j] = (kb == 0) ? p[j] : (_Float16)0.0f;
  return a;
}

// =======================  Kernel A: projections  =======================
// blockIdx.x = sample, blockIdx.y = matrix (0: Q=last@(Wqf+Wql)^T, 1: K=nodes@Wk^T, 2: VT)
__global__ __launch_bounds__(256) void proj_kernel(
    const float* __restrict__ nodes, const float* __restrict__ last,
    const float* __restrict__ Wqf, const float* __restrict__ Wql,
    const float* __restrict__ Wk,  const float* __restrict__ Wv,
    _Float16* __restrict__ Q, _Float16* __restrict__ K, _Float16* __restrict__ VT) {
  int b = blockIdx.x, mat = blockIdx.y;
  int wave = (int)(threadIdx.x >> 5), l = lane_id();
  int half = l >> 4, ln = l & 15;
  int nb = wave * 16;  // output col tile
  const float* X = (mat == 0) ? (last + b * NROW_ * EMB_) : (nodes + b * NROW_ * EMB_);

  v16h Bf[4];
#pragma unroll
  for (int ks = 0; ks < 4; ++ks) {
    int koff = ks * 32;
    if (mat == 0) {  // combined q weight: Wq = Wq_first + Wq_last
      int n = nb + ln, k0 = koff + half * 16;
      const float* p0 = Wqf + n * EMB_ + k0;
      const float* p1 = Wql + n * EMB_ + k0;
#pragma unroll
      for (int j = 0; j < 16; ++j) Bf[ks][j] = (_Float16)(p0[j] + p1[j]);
    } else {
      Bf[ks] = frag_b_wT_f32((mat == 1) ? Wk : Wv, EMB_, nb, koff);
    }
  }

  int mtiles = (mat == 2) ? 8 : MT_;  // VT padded to 128 node-rows
  for (int t = 0; t < mtiles; ++t) {
    v8f acc = {};
#pragma unroll
    for (int ks = 0; ks < 4; ++ks)
      acc = wmma16(frag_a_f32(X, EMB_, t * 16, NROW_, ks * 32), Bf[ks], acc);
    if (mat <= 1) {
      _Float16* O = ((mat == 0) ? Q : K) + b * NPAD_ * EMB_;
      int col = nb + ln;
#pragma unroll
      for (int i = 0; i < 8; ++i) O[(t * 16 + i + 8 * half) * EMB_ + col] = (_Float16)acc[i];
    } else {
      _Float16* O = VT + b * EMB_ * EMB_;
      int col = nb + ln;  // emb index -> VT row
#pragma unroll
      for (int i = 0; i < 8; ++i) O[col * EMB_ + (t * 16 + i + 8 * half)] = (_Float16)acc[i];
    }
  }
}

// =======================  Kernel B: attention + FFN + score  =======================
__global__ __launch_bounds__(256) void decoder_kernel(
    const float* __restrict__ nodes, const float* __restrict__ ninf,
    const float* __restrict__ newW,  const float* __restrict__ newB,
    const float* __restrict__ new2W, const float* __restrict__ new2B,
    const float* __restrict__ combW, const float* __restrict__ combB,
    const _Float16* __restrict__ Q, const _Float16* __restrict__ K,
    const _Float16* __restrict__ VT, float* __restrict__ out) {
  __shared__ __align__(16) _Float16 sOut[NPAD_ * EMB_];      // out_concat -> h2 (28 KB)
  __shared__ __align__(16) _Float16 sP[8 * 16 * EMB_];       // per-wave P -> h -> mh (32 KB)

  int b = blockIdx.x;
  int wave = (int)(threadIdx.x >> 5), l = lane_id();
  int half = l >> 4, ln = l & 15;
  const _Float16* Qb = Q + b * NPAD_ * EMB_;
  const _Float16* Kb = K + b * NPAD_ * EMB_;
  const _Float16* Vb = VT + b * EMB_ * EMB_;
  const float* nodesB = nodes + b * NROW_ * EMB_;
  const float* maskB  = ninf + b * NROW_ * NROW_;

  // ---------- Phase 1: masked MHA, wave = head ----------
  {
    int hb = wave * 16;
    v16h Bk[MT_], Bv[4];
#pragma unroll
    for (int nt = 0; nt < MT_; ++nt) Bk[nt] = frag_b_khead(Kb, nt * 16, hb);
#pragma unroll
    for (int ks = 0; ks < 4; ++ks) Bv[ks] = frag_b_wT_f16(Vb, EMB_, hb, ks * 32);
    _Float16* myP = sP + wave * 16 * EMB_;

    for (int t = 0; t < MT_; ++t) {
      v16h Aq = frag_a_qhead(Qb, t * 16, hb);
      v8f sc[MT_];
#pragma unroll
      for (int nt = 0; nt < MT_; ++nt) {
        v8f z = {};
        sc[nt] = wmma16(Aq, Bk[nt], z);
      }
      // scale (1/sqrt(16)) + mask + kill pad cols
#pragma unroll
      for (int nt = 0; nt < MT_; ++nt) {
        int n = nt * 16 + ln;
#pragma unroll
        for (int i = 0; i < 8; ++i) {
          int g = t * 16 + i + 8 * half;
          float s = sc[nt][i] * 0.25f;
          if (n < NROW_) { if (g < NROW_) s += maskB[g * NROW_ + n]; }
          else s = -1e30f;
          sc[nt][i] = s;
        }
      }
      // row softmax: rows striped -> reduce over 16-lane half + 7 tiles
      float mx[8], sm[8];
#pragma unroll
      for (int i = 0; i < 8; ++i) {
        float m = sc[0][i];
#pragma unroll
        for (int nt = 1; nt < MT_; ++nt) m = fmaxf(m, sc[nt][i]);
#pragma unroll
        for (int d = 1; d < 16; d <<= 1) m = fmaxf(m, __shfl_xor(m, d, 32));
        mx[i] = m; sm[i] = 0.0f;
      }
#pragma unroll
      for (int nt = 0; nt < MT_; ++nt)
#pragma unroll
        for (int i = 0; i < 8; ++i) {
          float e = __expf(sc[nt][i] - mx[i]);
          sc[nt][i] = e; sm[i] += e;
        }
#pragma unroll
      for (int i = 0; i < 8; ++i) {
#pragma unroll
        for (int d = 1; d < 16; d <<= 1) sm[i] += __shfl_xor(sm[i], d, 32);
        sm[i] = 1.0f / sm[i];
      }
      // stage P (f16) in per-wave LDS slab; zero pad cols 112..127
#pragma unroll
      for (int nt = 0; nt < MT_; ++nt)
#pragma unroll
        for (int i = 0; i < 8; ++i)
          myP[(i + 8 * half) * EMB_ + nt * 16 + ln] = (_Float16)(sc[nt][i] * sm[i]);
#pragma unroll
      for (int i = 0; i < 8; ++i)
        myP[(i + 8 * half) * EMB_ + NPAD_ + ln] = (_Float16)0.0f;
      __syncthreads();
      // out tile = P @ V_head
      v8f oa = {};
#pragma unroll
      for (int ks = 0; ks < 4; ++ks)
        oa = wmma16(frag_a_lds(myP, EMB_, 0, ks * 32), Bv[ks], oa);
#pragma unroll
      for (int i = 0; i < 8; ++i)
        sOut[(t * 16 + i + 8 * half) * EMB_ + hb + ln] = (_Float16)oa[i];
      __syncthreads();
    }
  }

  // ---------- Phase 2: h = relu(oc @ new[b] + bias) -> sP ----------
  {
    int e = wave * 16 + ln;
    const float* Wb = newW + b * EMB_ * EMB_;
    v16h Bn[4];
#pragma unroll
    for (int ks = 0; ks < 4; ++ks) Bn[ks] = frag_b_rm_f32(Wb, EMB_, wave * 16, ks * 32);
    float bias = newB[b * EMB_ + e];
    for (int t = 0; t < MT_; ++t) {
      v8f acc = {};
#pragma unroll
      for (int ks = 0; ks < 4; ++ks)
        acc = wmma16(frag_a_lds(sOut, EMB_, t * 16, ks * 32), Bn[ks], acc);
#pragma unroll
      for (int i = 0; i < 8; ++i) {
        float h = acc[i] + bias;
        sP[(t * 16 + i + 8 * half) * EMB_ + e] = (_Float16)(h > 0.0f ? h : 0.0f);
      }
    }
  }
  __syncthreads();

  // ---------- Phase 3: h2 = h @ new2[b] + bias2 + oc -> sOut ----------
  {
    int e = wave * 16 + ln;
    const float* Wb = new2W + b * EMB_ * EMB_;
    v16h Bn[4];
#pragma unroll
    for (int ks = 0; ks < 4; ++ks) Bn[ks] = frag_b_rm_f32(Wb, EMB_, wave * 16, ks * 32);
    float bias = new2B[b * EMB_ + e];
    for (int t = 0; t < MT_; ++t) {
      v8f acc = {};
#pragma unroll
      for (int ks = 0; ks < 4; ++ks)
        acc = wmma16(frag_a_lds(sP, EMB_, t * 16, ks * 32), Bn[ks], acc);
#pragma unroll
      for (int i = 0; i < 8; ++i) {
        int idx = (t * 16 + i + 8 * half) * EMB_ + e;  // wave-exclusive column
        float v = acc[i] + bias + (float)sOut[idx];
        sOut[idx] = (_Float16)v;
      }
    }
  }
  __syncthreads();

  // ---------- Phase 4: mh = h2 @ comb_W^T + comb_b -> sP ----------
  {
    int e = wave * 16 + ln;
    v16h Bc[4];
#pragma unroll
    for (int ks = 0; ks < 4; ++ks) Bc[ks] = frag_b_wT_f32(combW, EMB_, wave * 16, ks * 32);
    float cb = combB[e];
    for (int t = 0; t < MT_; ++t) {
      v8f acc = {};
#pragma unroll
      for (int ks = 0; ks < 4; ++ks)
        acc = wmma16(frag_a_lds(sOut, EMB_, t * 16, ks * 32), Bc[ks], acc);
#pragma unroll
      for (int i = 0; i < 8; ++i)
        sP[(t * 16 + i + 8 * half) * EMB_ + e] = (_Float16)(acc[i] + cb);
    }
  }
  __syncthreads();

  // ---------- Phase 5: score = clip*tanh(mh @ nodes^T / sqrt(E)) + mask; softmax ----------
  if (wave < MT_) {  // wave = row tile; whole-wave uniform branch (EXEC all ones inside)
    v16h Am[4];
#pragma unroll
    for (int ks = 0; ks < 4; ++ks) Am[ks] = frag_a_lds(sP, EMB_, wave * 16, ks * 32);
    v8f sc[MT_];
    for (int nt = 0; nt < MT_; ++nt) {
      v8f acc = {};
#pragma unroll
      for (int ks = 0; ks < 4; ++ks)
        acc = wmma16(Am[ks], frag_b_wT_f32_guard(nodesB, EMB_, nt * 16, NROW_, ks * 32), acc);
      sc[nt] = acc;
    }
#pragma unroll
    for (int nt = 0; nt < MT_; ++nt) {
      int n = nt * 16 + ln;
#pragma unroll
      for (int i = 0; i < 8; ++i) {
        int g = wave * 16 + i + 8 * half;
        float v = 10.0f * tanhf(sc[nt][i] * 0.08838834764831845f);  // 1/sqrt(128)
        if (n < NROW_) { if (g < NROW_) v += maskB[g * NROW_ + n]; }
        else v = -1e30f;
        sc[nt][i] = v;
      }
    }
    float mx[8], sm[8];
#pragma unroll
    for (int i = 0; i < 8; ++i) {
      float m = sc[0][i];
#pragma unroll
      for (int nt = 1; nt < MT_; ++nt) m = fmaxf(m, sc[nt][i]);
#pragma unroll
      for (int d = 1; d < 16; d <<= 1) m = fmaxf(m, __shfl_xor(m, d, 32));
      mx[i] = m; sm[i] = 0.0f;
    }
#pragma unroll
    for (int nt = 0; nt < MT_; ++nt)
#pragma unroll
      for (int i = 0; i < 8; ++i) {
        float e = __expf(sc[nt][i] - mx[i]);
        sc[nt][i] = e; sm[i] += e;
      }
#pragma unroll
    for (int i = 0; i < 8; ++i) {
#pragma unroll
      for (int d = 1; d < 16; d <<= 1) sm[i] += __shfl_xor(sm[i], d, 32);
      sm[i] = 1.0f / sm[i];
    }
#pragma unroll
    for (int nt = 0; nt < MT_; ++nt) {
      int n = nt * 16 + ln;
      if (n < NROW_) {
#pragma unroll
        for (int i = 0; i < 8; ++i) {
          int g = wave * 16 + i + 8 * half;
          if (g < NROW_)
            out[((long)b * NROW_ + g) * NROW_ + n] = sc[nt][i] * sm[i];
        }
      }
    }
  }
}

extern "C" void kernel_launch(void* const* d_in, const int* in_sizes, int n_in,
                              void* d_out, int out_size, void* d_ws, size_t ws_size,
                              hipStream_t stream) {
  const float* nodes = (const float*)d_in[0];
  const float* last  = (const float*)d_in[1];
  const float* ninf  = (const float*)d_in[2];
  const float* Wqf   = (const float*)d_in[3];
  const float* Wql   = (const float*)d_in[4];
  const float* Wk    = (const float*)d_in[5];
  const float* Wv    = (const float*)d_in[6];
  const float* newW  = (const float*)d_in[7];
  const float* newB  = (const float*)d_in[8];
  const float* new2W = (const float*)d_in[9];
  const float* new2B = (const float*)d_in[10];
  const float* combW = (const float*)d_in[11];
  const float* combB = (const float*)d_in[12];
  float* outp = (float*)d_out;

  int B = in_sizes[0] / (NROW_ * EMB_);  // 1024
  _Float16* ws = (_Float16*)d_ws;
  _Float16* Q  = ws;
  _Float16* K  = Q + (size_t)B * NPAD_ * EMB_;
  _Float16* VT = K + (size_t)B * NPAD_ * EMB_;

  proj_kernel<<<dim3(B, 3), 256, 0, stream>>>(nodes, last, Wqf, Wql, Wk, Wv, Q, K, VT);
  decoder_kernel<<<dim3(B), 256, 0, stream>>>(nodes, ninf, newW, newB, new2W, new2B,
                                              combW, combB, Q, K, VT, outp);
}
